// RIMRouter_47699906789716
// MI455X (gfx1250) — matrix operand order repaired and checked by reading, not compiled, not run on hardware
//
#include <hip/hip_runtime.h>

// ---------------------------------------------------------------------------
// CDNA5 / gfx1250 RIM router.  bf16 WMMA (v_wmma_f32_16x16x32_bf16) for all
// GEMMs, f32 accumulate, persistent-scan kernel with LDS-resident state.
// ---------------------------------------------------------------------------

typedef __bf16 bf16;
typedef __attribute__((ext_vector_type(16))) __bf16 v16bf;
typedef __attribute__((ext_vector_type(8)))  __bf16 v8bf;
typedef __attribute__((ext_vector_type(8)))  float  v8f;

// problem dims (from reference)
static constexpr int kB = 8, kS = 2048, kH = 2048;
static constexpr int kE = 8, kK = 2, kR = 128;
static constexpr int kIK = 64, kIV = 400;
static constexpr int kCK = 32, kCV = 128, kCH = 4;

// padded GEMM shapes
static constexpr int KV_N  = 512;                 // IK+IV=464 padded to 512
static constexpr int KV_KC = kH / 32;             // 64 k-chunks
static constexpr int KV_NT = KV_N / 16;           // 32 n-tiles
static constexpr int G_N   = 4 * kR;              // 512 gate cols
static constexpr int G_K   = kIV + kR;            // 528 (pad to 544)
static constexpr int G_KC  = 17;                  // ceil(528/32)
static constexpr int G_NT  = G_N / 16;            // 32
static constexpr int A_N   = 128 + 128 + 512;     // q|k|v = 768
static constexpr int A_KC  = kR / 32;             // 4
static constexpr int A_NT  = A_N / 16;            // 48
static constexpr int O_N   = kR;                  // 128
static constexpr int O_KC  = (kCH * kCV) / 32;    // 16
static constexpr int O_NT  = O_N / 16;            // 8

// packed (fragment-major) bf16 weight sizes, in elements
static constexpr size_t PKV_TOT   = (size_t)KV_KC * KV_NT * 512;  // 1,048,576
static constexpr size_t PG_PER_E  = (size_t)G_KC  * G_NT  * 512;  //   278,528
static constexpr size_t PQ_PER_E  = (size_t)A_KC  * A_NT  * 512;  //    98,304
static constexpr size_t PO_PER_E  = (size_t)O_KC  * O_NT  * 512;  //    65,536
static constexpr size_t PACK_TOT  = PKV_TOT + 8*PG_PER_E + 8*PQ_PER_E + 8*PO_PER_E;

__device__ __forceinline__ bf16 f2bf(float f) { return (bf16)f; }
__device__ __forceinline__ float sigm(float x) { return 1.f / (1.f + __expf(-x)); }

// A fragment per ISA 16-bit A 16x32 layout:
// lanes 0-15 : row=lane,    elems 0-7 -> K kb+0..7,  elems 8-15 -> K kb+16..23
// lanes 16-31: row=lane-16, elems 0-7 -> K kb+8..15, elems 8-15 -> K kb+24..31
__device__ __forceinline__ v16bf load_a_frag(const bf16* As, int lda, int lane, int kbase) {
  int row  = lane & 15;
  int half = lane >> 4;
  const bf16* p = As + row * lda + kbase + half * 8;
  v8bf lo = *(const v8bf*)p;
  v8bf hi = *(const v8bf*)(p + 16);
  return __builtin_shufflevector(lo, hi, 0,1,2,3,4,5,6,7,8,9,10,11,12,13,14,15);
}

// B fragment: packed layout -> each lane reads 16 contiguous bf16 (32B)
__device__ __forceinline__ v16bf load_b_frag(const bf16* packed, int kc, int nt, int NT, int lane) {
  return *(const v16bf*)(packed + ((size_t)(kc * NT + nt) * 32 + lane) * 16);
}

__device__ __forceinline__ v8f wmma_bf16(v16bf a, v16bf b, v8f c) {
  return __builtin_amdgcn_wmma_f32_16x16x32_bf16(false, a, false, b, (short)0, c, false, false);
}

// device-scope grid barrier (8 co-resident blocks) with s_sleep backoff
__device__ __forceinline__ void grid_barrier(int* cnt, int* gen, int nblk) {
  __syncthreads();
  __threadfence();
  if (threadIdx.x == 0) {
    int g = __hip_atomic_load(gen, __ATOMIC_ACQUIRE, __HIP_MEMORY_SCOPE_AGENT);
    int prev = __hip_atomic_fetch_add(cnt, 1, __ATOMIC_ACQ_REL, __HIP_MEMORY_SCOPE_AGENT);
    if (prev == nblk - 1) {
      __hip_atomic_store(cnt, 0, __ATOMIC_RELAXED, __HIP_MEMORY_SCOPE_AGENT);
      __hip_atomic_fetch_add(gen, 1, __ATOMIC_ACQ_REL, __HIP_MEMORY_SCOPE_AGENT);
    } else {
      while (__hip_atomic_load(gen, __ATOMIC_ACQUIRE, __HIP_MEMORY_SCOPE_AGENT) == g)
        __builtin_amdgcn_s_sleep(1);
    }
  }
  __syncthreads();
  __threadfence();
}

// ---------------------------------------------------------------------------
// Kernel 1: repack all weights to fragment-major bf16 (run once per launch)
// ---------------------------------------------------------------------------
__global__ __launch_bounds__(256) void pack_weights(
    const float* __restrict__ Wk,  const float* __restrict__ Wv,
    const float* __restrict__ Wx2h,const float* __restrict__ Wh2h,
    const float* __restrict__ Wq_, const float* __restrict__ Wk_,
    const float* __restrict__ Wv_, const float* __restrict__ Wout,
    bf16* __restrict__ pKV, bf16* __restrict__ pG,
    bf16* __restrict__ pQKV, bf16* __restrict__ pOut)
{
  for (size_t idx = (size_t)blockIdx.x * 256 + threadIdx.x; idx < PACK_TOT;
       idx += (size_t)gridDim.x * 256) {
    if (idx < PKV_TOT) {                                   // Wk|Wv, K=2048,N=512
      int rem = (int)idx;
      int i = rem & 15, lane = (rem >> 4) & 31, q = rem >> 9;
      int nt = q & 31, kc = q >> 5;
      int K = kc * 32 + (lane >> 4) * 8 + (i & 7) + (i >> 3) * 16, col = nt * 16 + (lane & 15);
      float v = 0.f;
      if (col < kIK)            v = Wk[(size_t)K * kIK + col];
      else if (col < kIK + kIV) v = Wv[(size_t)K * kIV + (col - kIK)];
      pKV[rem] = f2bf(v);
    } else if (idx < PKV_TOT + 8 * PG_PER_E) {             // Wx2h|Wh2h, K=544,N=512
      size_t rem = idx - PKV_TOT;
      int e = (int)(rem / PG_PER_E); int r = (int)(rem % PG_PER_E);
      int i = r & 15, lane = (r >> 4) & 31, q = r >> 9;
      int nt = q & 31, kc = q >> 5;
      int K = kc * 32 + (lane >> 4) * 8 + (i & 7) + (i >> 3) * 16, col = nt * 16 + (lane & 15);
      float v = 0.f;
      if (K < kIV)      v = Wx2h[((size_t)e * kIV + K) * G_N + col];
      else if (K < G_K) v = Wh2h[((size_t)e * kR + (K - kIV)) * G_N + col];
      pG[(size_t)e * PG_PER_E + r] = f2bf(v);
    } else if (idx < PKV_TOT + 8 * PG_PER_E + 8 * PQ_PER_E) { // Wq_|Wk_|Wv_, K=128,N=768
      size_t rem = idx - PKV_TOT - 8 * PG_PER_E;
      int e = (int)(rem / PQ_PER_E); int r = (int)(rem % PQ_PER_E);
      int i = r & 15, lane = (r >> 4) & 31, q = r >> 9;
      int nt = q % A_NT, kc = q / A_NT;
      int K = kc * 32 + (lane >> 4) * 8 + (i & 7) + (i >> 3) * 16, col = nt * 16 + (lane & 15);
      float v;
      if (col < 128)      v = Wq_[((size_t)e * kR + K) * 128 + col];
      else if (col < 256) v = Wk_[((size_t)e * kR + K) * 128 + (col - 128)];
      else                v = Wv_[((size_t)e * kR + K) * 512 + (col - 256)];
      pQKV[(size_t)e * PQ_PER_E + r] = f2bf(v);
    } else {                                               // Wout, K=512,N=128
      size_t rem = idx - PKV_TOT - 8 * PG_PER_E - 8 * PQ_PER_E;
      int e = (int)(rem / PO_PER_E); int r = (int)(rem % PO_PER_E);
      int i = r & 15, lane = (r >> 4) & 31, q = r >> 9;
      int nt = q & 7, kc = q >> 3;
      int K = kc * 32 + (lane >> 4) * 8 + (i & 7) + (i >> 3) * 16, col = nt * 16 + (lane & 15);
      float v = Wout[((size_t)e * 512 + K) * 128 + col];
      pOut[(size_t)e * PO_PER_E + r] = f2bf(v);
    }
  }
}

// ---------------------------------------------------------------------------
// Kernel 2: hoisted input projections  kv[t][b][0:464] = x[b,t,:]@[Wk|Wv]+bias
// one block = 16 timesteps x 512 cols (8 waves x 4 wmma tiles), K=2048
// ---------------------------------------------------------------------------
__global__ __launch_bounds__(256) void kv_gemm(
    const float* __restrict__ x, const bf16* __restrict__ pKV,
    const float* __restrict__ bk, const float* __restrict__ bv,
    float* __restrict__ kv)
{
  __shared__ __align__(16) bf16 As[16 * 32];
  const int b  = blockIdx.x >> 7;          // 8 batches x 128 t-tiles
  const int t0 = (blockIdx.x & 127) << 4;
  const int wave = threadIdx.x >> 5, lane = threadIdx.x & 31;
  const float* xrow = x + ((size_t)b * kS + t0) * kH;

  v8f acc[4] = {};
  for (int kc = 0; kc < KV_KC; ++kc) {
    __syncthreads();
    int e0 = threadIdx.x * 2;              // 512 elems, 2 per thread
    int r = e0 >> 5, c = e0 & 31;
    As[e0]     = f2bf(xrow[(size_t)r * kH + kc * 32 + c]);
    As[e0 + 1] = f2bf(xrow[(size_t)r * kH + kc * 32 + c + 1]);
    __syncthreads();
    v16bf a = load_a_frag(As, 32, lane, 0);
#pragma unroll
    for (int i = 0; i < 4; ++i) {
      int nt = wave * 4 + i;
      if (kc + 1 < KV_KC)
        __builtin_prefetch(pKV + ((size_t)((kc + 1) * KV_NT + nt) * 32 + lane) * 16, 0, 1);
      acc[i] = wmma_bf16(a, load_b_frag(pKV, kc, nt, KV_NT, lane), acc[i]);
    }
  }
#pragma unroll
  for (int i = 0; i < 4; ++i) {
    int col = wave * 64 + i * 16 + (lane & 15);
    float bias = (col < kIK) ? bk[col] : ((col < kIK + kIV) ? bv[col - kIK] : 0.f);
    int rbase = (lane >> 4) * 8;
#pragma unroll
    for (int rr = 0; rr < 8; ++rr) {
      int t = t0 + rbase + rr;
      kv[((size_t)t * kB + b) * KV_N + col] = acc[i][rr] + bias;
    }
  }
}

// ---------------------------------------------------------------------------
// Kernel 3: persistent scan.  8 blocks (one per expert), 256 threads (8 waves).
// h,c live in LDS for all 2048 steps; 3 grid barriers / step.
// ---------------------------------------------------------------------------
__global__ __launch_bounds__(256) void rim_scan(
    const float* __restrict__ kv,
    const bf16* __restrict__ pG, const bf16* __restrict__ pQKV,
    const bf16* __restrict__ pOut,
    const float* __restrict__ Wq, const float* __restrict__ bk,
    const float* __restrict__ bv,
    const float* __restrict__ h0, const float* __restrict__ c0,
    float* g_s0, float* g_p0, float* g_k, float* g_v,
    int* bar_cnt, int* bar_gen,
    float* __restrict__ out_w, int* __restrict__ out_i)
{
  const int e = blockIdx.x;
  const int tid = threadIdx.x;
  const int wave = tid >> 5, lane = tid & 31;

  __shared__ float h_s[kB][kR];
  __shared__ float c_s[kB][kR];
  __shared__ float hn_s[kB][kR];
  __shared__ float qa_s[kB][kR];            // comm-attention q (CH*CK)
  __shared__ float q_s[kB][kIK];
  __shared__ float p0_s[kB], p1_s[kB];
  __shared__ float mask_s[kB][kE];
  __shared__ float att_s[kB][kCH][kE];
  __shared__ __align__(16) unsigned char pool[17408 + 16384]; // reused scratch
  bf16*  Ag    = (bf16*)pool;               // 16 x 544 bf16 gates A
  float* gates = (float*)(pool + 17408);    //  8 x 512 f32
  bf16*  Aa    = (bf16*)pool;               // 16 x 128 bf16 (h_new)
  bf16*  Ac    = (bf16*)pool;               // 16 x 512 bf16 (ctx)

  const bf16* pGe = pG   + (size_t)e * PG_PER_E;
  const bf16* pQe = pQKV + (size_t)e * PQ_PER_E;
  const bf16* pOe = pOut + (size_t)e * PO_PER_E;

  for (int i = tid; i < kB * kR; i += 256) {
    int b = i >> 7, d = i & 127;
    h_s[b][d] = h0[((size_t)b * kE + e) * kR + d];
    c_s[b][d] = c0[((size_t)b * kE + e) * kR + d];
  }
  __syncthreads();

  for (int t = 0; t < kS; ++t) {
    const float* kvt = kv + (size_t)t * kB * KV_N;

    // --- phase 0: routing q = h@Wq[e], scores, 2-way softmax -------------
    for (int o = tid; o < kB * kIK; o += 256) {
      int b = o >> 6, d = o & 63;
      const float* wq = Wq + ((size_t)e * kR) * kIK + d;
      float acc = 0.f;
      for (int r = 0; r < kR; ++r) acc += h_s[b][r] * wq[(size_t)r * kIK];
      q_s[b][d] = acc;
    }
    __syncthreads();
    if (tid < kB) {
      int b = tid;
      float s0 = 0.f, s1 = 0.f;
      for (int d = 0; d < kIK; ++d) {
        s0 += q_s[b][d] * kvt[b * KV_N + d];
        s1 += q_s[b][d] * bk[d];
      }
      s0 *= 0.125f; s1 *= 0.125f;                 // 1/sqrt(IK)
      float p0 = 1.f / (1.f + __expf(s1 - s0));
      p0_s[b] = p0; p1_s[b] = 1.f - p0;
      g_s0[b * kE + e] = s0;
      g_p0[b * kE + e] = p0;
    }
    grid_barrier(bar_cnt, bar_gen, kE);

    // --- phase 1: top-2 over experts, mask, emit routing outputs ----------
    if (tid < kB) {
      int b = tid;
      float sc[kE];
      for (int j = 0; j < kE; ++j) sc[j] = ((const volatile float*)g_s0)[b * kE + j];
      int i0 = 0; float v0 = sc[0];
      for (int j = 1; j < kE; ++j) if (sc[j] > v0) { v0 = sc[j]; i0 = j; }
      int i1 = (i0 == 0) ? 1 : 0; float v1 = sc[i1];
      for (int j = 0; j < kE; ++j) if (j != i0 && sc[j] > v1) { v1 = sc[j]; i1 = j; }
      for (int j = 0; j < kE; ++j) mask_s[b][j] = 0.f;
      mask_s[b][i0] = 1.f; mask_s[b][i1] = 1.f;
      if (e == 0) {
        int lo = i0 < i1 ? i0 : i1, hi = i0 < i1 ? i1 : i0;
        float plo = ((const volatile float*)g_p0)[b * kE + lo];
        float phi = ((const volatile float*)g_p0)[b * kE + hi];
        float inv = 1.f / (plo + phi);
        size_t o = ((size_t)t * kB + b) * kK;
        out_w[o] = plo * inv; out_w[o + 1] = phi * inv;
        out_i[o] = lo;        out_i[o + 1] = hi;
      }
    }
    __syncthreads();

    // --- phase 2: gates = [inp|h] @ [Wx2h;Wh2h]  (16x544x512 bf16 WMMA) ---
    for (int i2 = tid; i2 < 16 * 544; i2 += 256) {
      int r = i2 / 544, c = i2 % 544;
      float v = 0.f;
      if (r < kB) {
        if (c < kIV)      v = (p0_s[r] * kvt[r * KV_N + kIK + c] + p1_s[r] * bv[c]) * mask_s[r][e];
        else if (c < G_K) v = h_s[r][c - kIV];
      }
      Ag[i2] = f2bf(v);
    }
    __syncthreads();
    {
      v8f acc[4] = {};
      for (int kc = 0; kc < G_KC; ++kc) {
        v16bf a = load_a_frag(Ag, 544, lane, kc * 32);
#pragma unroll
        for (int i = 0; i < 4; ++i) {
          int nt = wave * 4 + i;
          if (kc + 1 < G_KC)
            __builtin_prefetch(pGe + ((size_t)((kc + 1) * G_NT + nt) * 32 + lane) * 16, 0, 1);
          acc[i] = wmma_bf16(a, load_b_frag(pGe, kc, nt, G_NT, lane), acc[i]);
        }
      }
      int rbase = (lane >> 4) * 8;
#pragma unroll
      for (int i = 0; i < 4; ++i) {
        int col = wave * 64 + i * 16 + (lane & 15);
#pragma unroll
        for (int rr = 0; rr < 8; ++rr)
          if (rbase + rr < kB) gates[(rbase + rr) * G_N + col] = acc[i][rr];
      }
    }
    __syncthreads();

    // --- phase 3: LSTM cell (f32) ----------------------------------------
    for (int i3 = tid; i3 < kB * kR; i3 += 256) {
      int b = i3 >> 7, d = i3 & 127;
      float ig = gates[b * G_N + d];
      float fg = gates[b * G_N + 128 + d];
      float gg = gates[b * G_N + 256 + d];
      float og = gates[b * G_N + 384 + d];
      float cn = sigm(fg) * c_s[b][d] + sigm(ig) * tanhf(gg);
      hn_s[b][d] = sigm(og) * tanhf(cn);
      if (mask_s[b][e] != 0.f) c_s[b][d] = cn;    // c_out = mask?c_new:c
    }
    __syncthreads();

    // --- phase 4: comm q|k|v = h_new @ [Wq_|Wk_|Wv_]  (16x128x768) --------
    for (int i4 = tid; i4 < 16 * kR; i4 += 256) {
      int r = i4 >> 7, d = i4 & 127;
      Aa[i4] = f2bf(r < kB ? hn_s[r][d] : 0.f);
    }
    __syncthreads();
    {
      v8f acc[6] = {};
      for (int kc = 0; kc < A_KC; ++kc) {
        v16bf a = load_a_frag(Aa, kR, lane, kc * 32);
#pragma unroll
        for (int i = 0; i < 6; ++i)
          acc[i] = wmma_bf16(a, load_b_frag(pQe, kc, wave * 6 + i, A_NT, lane), acc[i]);
      }
      int rbase = (lane >> 4) * 8;
#pragma unroll
      for (int i = 0; i < 6; ++i) {
        int col = (wave * 6 + i) * 16 + (lane & 15);
#pragma unroll
        for (int rr = 0; rr < 8; ++rr) {
          int row = rbase + rr;
          if (row < kB) {
            float v = acc[i][rr];
            if (col < 128)      qa_s[row][col] = v;
            else if (col < 256) g_k[((size_t)e * kB + row) * kR + (col - 128)] = v;
            else                g_v[((size_t)e * kB + row) * 512 + (col - 256)] = v;
          }
        }
      }
    }
    grid_barrier(bar_cnt, bar_gen, kE);

    // --- phase 5: attention over experts (tiny, VALU) ---------------------
    {
      int b = tid >> 5, hh = (tid >> 3) & 3, f = tid & 7;  // 8*4*8 = 256
      const volatile float* kp = (const volatile float*)g_k + ((size_t)f * kB + b) * kR + hh * kCK;
      float s = 0.f;
      for (int j = 0; j < kCK; ++j) s += qa_s[b][hh * kCK + j] * kp[j];
      att_s[b][hh][f] = s * 0.1767766952966369f;           // 1/sqrt(CK)
    }
    __syncthreads();
    if (tid < kB * kCH) {
      int b = tid >> 2, hh = tid & 3;
      float m = att_s[b][hh][0];
      for (int f = 1; f < kE; ++f) m = fmaxf(m, att_s[b][hh][f]);
      float ex[kE], sum = 0.f;
      for (int f = 0; f < kE; ++f) { ex[f] = __expf(att_s[b][hh][f] - m); sum += ex[f]; }
      float inv = mask_s[b][e] / sum;                      // fold mask into softmax
      for (int f = 0; f < kE; ++f) att_s[b][hh][f] = ex[f] * inv;
    }
    __syncthreads();
    for (int i5 = tid; i5 < 16 * 512; i5 += 256) {         // ctx -> bf16 A
      int r = i5 >> 9, c = i5 & 511;
      float v = 0.f;
      if (r < kB) {
        int hh = c >> 7, d = c & 127;
        const volatile float* vp = (const volatile float*)g_v;
        for (int f = 0; f < kE; ++f)
          v += att_s[r][hh][f] * vp[((size_t)f * kB + r) * 512 + hh * kCV + d];
      }
      Ac[i5] = f2bf(v);
    }
    __syncthreads();

    // --- phase 6: h_comm = ctx@Wout + h_new ; masked state update ---------
    {
      v8f acc = {};
      for (int kc = 0; kc < O_KC; ++kc) {
        v16bf a = load_a_frag(Ac, 512, lane, kc * 32);
        acc = wmma_bf16(a, load_b_frag(pOe, kc, wave, O_NT, lane), acc);
      }
      int col = wave * 16 + (lane & 15);
      int rbase = (lane >> 4) * 8;
#pragma unroll
      for (int rr = 0; rr < 8; ++rr) {
        int row = rbase + rr;
        if (row < kB && mask_s[row][e] != 0.f)
          h_s[row][col] = acc[rr] + hn_s[row][col];
      }
    }
    grid_barrier(bar_cnt, bar_gen, kE);   // protects reuse of g_* next step
  }
}

// ---------------------------------------------------------------------------
// Host launcher.  Workspace layout (all 256B-aligned, ~41 MB total):
//   [barrier 256B][g_s0 256B][g_p0 256B][g_k 32KB][g_v 128KB]
//   [pKV 2MB][pG 4.25MB][pQKV 1.5MB][pOut 1MB][kv 32MB]
// ---------------------------------------------------------------------------
extern "C" void kernel_launch(void* const* d_in, const int* in_sizes, int n_in,
                              void* d_out, int out_size, void* d_ws, size_t ws_size,
                              hipStream_t stream) {
  (void)in_sizes; (void)n_in; (void)out_size; (void)ws_size;
  const float* x    = (const float*)d_in[0];
  const float* h0   = (const float*)d_in[1];
  const float* c0   = (const float*)d_in[2];
  const float* Wk   = (const float*)d_in[3];
  const float* bk   = (const float*)d_in[4];
  const float* Wv   = (const float*)d_in[5];
  const float* bv   = (const float*)d_in[6];
  const float* Wq   = (const float*)d_in[7];
  const float* Wx2h = (const float*)d_in[8];
  const float* Wh2h = (const float*)d_in[9];
  const float* Wq_  = (const float*)d_in[10];
  const float* Wk_  = (const float*)d_in[11];
  const float* Wv_  = (const float*)d_in[12];
  const float* Wout = (const float*)d_in[13];

  char* cur = (char*)d_ws;
  int*   bar   = (int*)cur;            cur += 256;
  float* g_s0  = (float*)cur;          cur += 256;
  float* g_p0  = (float*)cur;          cur += 256;
  float* g_k   = (float*)cur;          cur += (size_t)kE * kB * kR * 4;        // 32KB
  float* g_v   = (float*)cur;          cur += (size_t)kE * kB * 512 * 4;       // 128KB
  bf16*  pKV   = (bf16*)cur;           cur += PKV_TOT * 2;
  bf16*  pG    = (bf16*)cur;           cur += 8 * PG_PER_E * 2;
  bf16*  pQKV  = (bf16*)cur;           cur += 8 * PQ_PER_E * 2;
  bf16*  pOut  = (bf16*)cur;           cur += 8 * PO_PER_E * 2;
  float* kv    = (float*)cur;          cur += (size_t)kS * kB * KV_N * 4;      // 32MB

  float* out_w = (float*)d_out;
  int*   out_i = (int*)d_out + (size_t)kS * kB * kK;

  hipMemsetAsync(bar, 0, 256, stream);

  int packBlocks = (int)((PACK_TOT + 255) / 256);
  pack_weights<<<packBlocks, 256, 0, stream>>>(Wk, Wv, Wx2h, Wh2h, Wq_, Wk_, Wv_, Wout,
                                               pKV, pG, pQKV, pOut);
  kv_gemm<<<kB * (kS / 16), 256, 0, stream>>>(x, pKV, bk, bv, kv);
  rim_scan<<<kE, 256, 0, stream>>>(kv, pG, pQKV, pOut, Wq, bk, bv, h0, c0,
                                   g_s0, g_p0, g_k, g_v, bar, bar + 1, out_w, out_i);
}
